// TwoPhasePinn_41059887350380
// MI455X (gfx1250) — matrix operand order repaired and checked by reading, not compiled, not run on hardware
//
#include <hip/hip_runtime.h>
#include <hip/hip_bf16.h>

typedef _Float16 half_t;
typedef __attribute__((ext_vector_type(16))) _Float16     v16h;
typedef __attribute__((ext_vector_type(8)))  float        v8f;
typedef __attribute__((ext_vector_type(4)))  unsigned int v4u;

#define NPTS   32768
#define HDIM   250
#define HP     256
#define TP     16            // points per block
#define ROWS   (TP * 10)     // 160 extended rows (value + 3 jac + 6 hess)
#define SA     264           // ACT LDS row stride in halves (bank-skewed)
#define SW     264           // weight LDS row stride in halves
#define NLAYERS 6
#define ACT_BYTES  (ROWS * SA * 2)       // 84480
#define WB_BYTES   (HP * SW * 2)         // 135168
#define SMEM_BYTES (ACT_BYTES + WB_BYTES)

#if defined(__gfx1250__)
#define ASYNC_LDS 1
typedef __attribute__((address_space(3))) char lds_char;
#else
#define ASYNC_LDS 0
#endif

// 16-byte global -> LDS async copy (per-lane addresses), tracked by ASYNCcnt.
__device__ __forceinline__ void async_copy16(const char* gsrc, char* ldst) {
#if ASYNC_LDS
    unsigned lds = (unsigned)(unsigned long long)(lds_char*)ldst;
    unsigned long long ga = (unsigned long long)gsrc;
    asm volatile("global_load_async_to_lds_b128 %0, %1, off"
                 :: "v"(lds), "v"(ga) : "memory");
#else
    *(v4u*)ldst = *(const v4u*)gsrc;
#endif
}

__device__ __forceinline__ void wait_async() {
#if ASYNC_LDS
#if __has_builtin(__builtin_amdgcn_s_wait_asynccnt)
    __builtin_amdgcn_s_wait_asynccnt(0);
#else
    asm volatile("s_wait_asynccnt 0" ::: "memory");
#endif
#endif
}

union Frag { v16h h; v4u u[2]; };

// ---------------------------------------------------------------------------
// Prep: convert W_h (6 x 250 x 250 fp32, [k][n]) into padded f16, transposed
// to [l][n][k] so B-fragment K-reads are contiguous.  Zero padding rows/cols.
// ---------------------------------------------------------------------------
__global__ void prep_weights(const float* __restrict__ W_h, half_t* __restrict__ Wp) {
    int idx = blockIdx.x * blockDim.x + threadIdx.x;
    if (idx >= NLAYERS * HP * HP) return;
    int l = idx >> 16;
    int r = idx & 0xFFFF;
    int n = r >> 8;
    int k = r & 255;
    float w = (n < HDIM && k < HDIM) ? W_h[l * HDIM * HDIM + k * HDIM + n] : 0.0f;
    Wp[idx] = (half_t)w;
}

// ---------------------------------------------------------------------------
// tanh nonlinearity with 2nd-order forward-mode chain rule.
// Thread owns column n = tid across all 16 points (no races).
// z channels: 0=val 1=x 2=y 3=t 4=xx 5=xy 6=xt 7=yy 8=yt 9=tt
// ---------------------------------------------------------------------------
__device__ __forceinline__ void apply_nonlin(half_t* ACT, const float* __restrict__ bias,
                                             float cc, int tid) {
    const int n = tid;
    const float bv = (n < HDIM) ? bias[n] : 0.0f;
    for (int p = 0; p < TP; ++p) {
        int base = (p * 10) * SA + n;
        float z0  = (float)ACT[base]        + bv;
        float zx  = (float)ACT[base + 1*SA];
        float zy  = (float)ACT[base + 2*SA];
        float zt  = (float)ACT[base + 3*SA];
        float zxx = (float)ACT[base + 4*SA];
        float zxy = (float)ACT[base + 5*SA];
        float zxt = (float)ACT[base + 6*SA];
        float zyy = (float)ACT[base + 7*SA];
        float zyt = (float)ACT[base + 8*SA];
        float ztt = (float)ACT[base + 9*SA];
        float a  = tanhf(cc * z0);
        float s  = 1.0f - a * a;
        float f1 = cc * s;
        float f2 = -2.0f * cc * cc * a * s;
        ACT[base        ] = (half_t)a;
        ACT[base + 1*SA ] = (half_t)(f1 * zx);
        ACT[base + 2*SA ] = (half_t)(f1 * zy);
        ACT[base + 3*SA ] = (half_t)(f1 * zt);
        ACT[base + 4*SA ] = (half_t)(f1 * zxx + f2 * zx * zx);
        ACT[base + 5*SA ] = (half_t)(f1 * zxy + f2 * zx * zy);
        ACT[base + 6*SA ] = (half_t)(f1 * zxt + f2 * zx * zt);
        ACT[base + 7*SA ] = (half_t)(f1 * zyy + f2 * zy * zy);
        ACT[base + 8*SA ] = (half_t)(f1 * zyt + f2 * zy * zt);
        ACT[base + 9*SA ] = (half_t)(f1 * ztt + f2 * zt * zt);
    }
}

// ---------------------------------------------------------------------------
// Fused PINN kernel: 16 points/block, 8 wave32 waves.
// ---------------------------------------------------------------------------
__global__ __launch_bounds__(256, 1)
void pinn_fused(const float* __restrict__ x, const float* __restrict__ y,
                const float* __restrict__ t,
                const float* __restrict__ W_in, const float* __restrict__ b_in,
                const float* __restrict__ b_h,
                const float* __restrict__ W_out, const float* __restrict__ b_out,
                const float* __restrict__ act_coeff,
                const half_t* __restrict__ Wp,
                float* __restrict__ out) {
    extern __shared__ char smem[];
    half_t* ACT = (half_t*)smem;
    half_t* WB  = (half_t*)(smem + ACT_BYTES);

    const int tid  = threadIdx.x;
    const int lane = tid & 31;
    const int wave = tid >> 5;
    const int p0   = blockIdx.x * TP;
    const int lhalf = (lane >> 4) & 1;    // 0 for lanes 0-15, 1 for 16-31

    // ---- Stage A: input layer (3 -> 250) + tanh; seed tangents with W_in rows
    {
        const float c0 = 10.0f * act_coeff[0];
        const int n = tid;
        float w0 = 0.f, w1 = 0.f, w2 = 0.f, bi = 0.f;
        if (n < HDIM) { w0 = W_in[n]; w1 = W_in[HDIM + n]; w2 = W_in[2 * HDIM + n]; bi = b_in[n]; }
        for (int p = 0; p < TP; ++p) {
            float a = 0.f, f1 = 0.f, f2 = 0.f;
            if (n < HDIM) {
                float z = x[p0 + p] * w0 + y[p0 + p] * w1 + t[p0 + p] * w2 + bi;
                a = tanhf(c0 * z);
                float s = 1.0f - a * a;
                f1 = c0 * s;
                f2 = -2.0f * c0 * c0 * a * s;
            }
            int base = (p * 10) * SA + n;
            ACT[base        ] = (half_t)a;
            ACT[base + 1*SA ] = (half_t)(f1 * w0);
            ACT[base + 2*SA ] = (half_t)(f1 * w1);
            ACT[base + 3*SA ] = (half_t)(f1 * w2);
            ACT[base + 4*SA ] = (half_t)(f2 * w0 * w0);
            ACT[base + 5*SA ] = (half_t)(f2 * w0 * w1);
            ACT[base + 6*SA ] = (half_t)(f2 * w0 * w2);
            ACT[base + 7*SA ] = (half_t)(f2 * w1 * w1);
            ACT[base + 8*SA ] = (half_t)(f2 * w1 * w2);
            ACT[base + 9*SA ] = (half_t)(f2 * w2 * w2);
        }
    }

    // ---- Stage B: 6 hidden layers, WMMA GEMM (160x256 @ 256x256)
    for (int l = 0; l < NLAYERS; ++l) {
        // Stage this layer's f16 weights into LDS (async; overlaps nonlinearity).
        {
            const char* gsrc = (const char*)(Wp + (size_t)l * HP * HP);
            char* wbase = (char*)WB;
            for (int i = 0; i < 32; ++i) {
                int u = tid + 256 * i;           // 8192 x 16B chunks
                int n = u >> 5, c = u & 31;
                int goff = n * 512 + c * 16;     // source: tight 512B rows
                int loff = n * (SW * 2) + c * 16;// dest: 528B padded rows
                async_copy16(gsrc + goff, wbase + loff);
            }
        }
        // Pointwise nonlinearity on previous layer's raw z (overlaps async copy).
        if (l > 0) apply_nonlin(ACT, b_h + (l - 1) * HDIM, 10.0f * act_coeff[l], tid);
        wait_async();
        __syncthreads();

        // GEMM: each wave owns N-tiles {2w, 2w+1} x all 10 M-tiles.
        const int nt0 = wave * 2, nt1 = nt0 + 1;
        v8f acc0[10], acc1[10];
#pragma unroll
        for (int m = 0; m < 10; ++m) {
            acc0[m] = (v8f){0.f,0.f,0.f,0.f,0.f,0.f,0.f,0.f};
            acc1[m] = (v8f){0.f,0.f,0.f,0.f,0.f,0.f,0.f,0.f};
        }
#pragma unroll
        for (int kk = 0; kk < 8; ++kk) {
            // B fragments (32x16 f16): lane<16 -> K 0..15, lane>=16 -> K 16..31
            Frag b0, b1;
            int kb = kk * 32 + lhalf * 16;
            int nr0 = (nt0 * 16 + (lane & 15)) * SW;
            int nr1 = (nt1 * 16 + (lane & 15)) * SW;
            b0.u[0] = *(const v4u*)(WB + nr0 + kb);
            b0.u[1] = *(const v4u*)(WB + nr0 + kb + 8);
            b1.u[0] = *(const v4u*)(WB + nr1 + kb);
            b1.u[1] = *(const v4u*)(WB + nr1 + kb + 8);
#pragma unroll
            for (int m = 0; m < 10; ++m) {
                // A fragment (16x32 f16): lanes 0-15 K 0-15, lanes 16-31 K 16-31
                Frag aF;
                int ra = (m * 16 + (lane & 15)) * SA;
                int ka = kk * 32 + lhalf * 16;
                aF.u[0] = *(const v4u*)(ACT + ra + ka);
                aF.u[1] = *(const v4u*)(ACT + ra + ka + 8);
                acc0[m] = __builtin_amdgcn_wmma_f32_16x16x32_f16(
                    false, aF.h, false, b0.h, (short)0, acc0[m], false, false);
                acc1[m] = __builtin_amdgcn_wmma_f32_16x16x32_f16(
                    false, aF.h, false, b1.h, (short)0, acc1[m], false, false);
            }
        }
        __syncthreads();   // all A-reads of ACT complete before overwrite

        // Store raw z back to ACT (f16).  C layout: VGPR g -> row g (+8 for hi lanes).
#pragma unroll
        for (int m = 0; m < 10; ++m) {
            int rbase = m * 16 + lhalf * 8;
            int col0 = nt0 * 16 + (lane & 15);
            int col1 = nt1 * 16 + (lane & 15);
#pragma unroll
            for (int g = 0; g < 8; ++g) {
                ACT[(rbase + g) * SA + col0] = (half_t)acc0[m][g];
                ACT[(rbase + g) * SA + col1] = (half_t)acc1[m][g];
            }
        }
        __syncthreads();
    }

    // Final nonlinearity (layer 5 output uses act_coeff[6], b_h[5])
    apply_nonlin(ACT, b_h + (NLAYERS - 1) * HDIM, 10.0f * act_coeff[NLAYERS], tid);
    __syncthreads();

    // ---- Stage C: output layer 250 -> 4 for all 10 channels (VALU, tiny)
    float* ZO = (float*)WB;   // weight buffer is dead now; reuse for 16x10x4 f32
    if (tid < ROWS) {
        float s0 = 0.f, s1 = 0.f, s2 = 0.f, s3 = 0.f;
        const half_t* ar = ACT + tid * SA;   // row == p*10 + c == tid
        for (int k = 0; k < HDIM; ++k) {
            float h = (float)ar[k];
            s0 += h * W_out[k * 4 + 0];
            s1 += h * W_out[k * 4 + 1];
            s2 += h * W_out[k * 4 + 2];
            s3 += h * W_out[k * 4 + 3];
        }
        if ((tid % 10) == 0) { s0 += b_out[0]; s1 += b_out[1]; s2 += b_out[2]; s3 += b_out[3]; }
        ZO[tid * 4 + 0] = s0; ZO[tid * 4 + 1] = s1;
        ZO[tid * 4 + 2] = s2; ZO[tid * 4 + 3] = s3;
    }
    __syncthreads();

    // ---- Stage D: output transforms + PDE residuals (1 thread / point)
    if (tid < TP) {
        const float* zo = ZO + tid * 40;
#define Z(c, j) zo[(c) * 4 + (j)]
        float u    = Z(0, 0), v    = Z(0, 1);
        float z2   = Z(0, 2), z3   = Z(0, 3);
        float u_x  = Z(1, 0), u_y  = Z(2, 0), u_t = Z(3, 0);
        float v_x  = Z(1, 1), v_y  = Z(2, 1), v_t = Z(3, 1);
        float u_xx = Z(4, 0), u_yy = Z(7, 0);
        float v_xx = Z(4, 1), v_yy = Z(7, 1);
        float ex   = expf(z2);
        float p_x  = ex * Z(1, 2), p_y = ex * Z(2, 2);
        float sg   = 1.0f / (1.0f + expf(-z3));
        float sp   = sg * (1.0f - sg);
        float spp  = sp * (1.0f - 2.0f * sg);
        float z3x = Z(1, 3), z3y = Z(2, 3), z3t = Z(3, 3);
        float a    = sg;
        float a_x  = sp * z3x, a_y = sp * z3y, a_t = sp * z3t;
        float a_xx = spp * z3x * z3x + sp * Z(4, 3);
        float a_xy = spp * z3x * z3y + sp * Z(5, 3);
        float a_yy = spp * z3y * z3y + sp * Z(7, 3);
#undef Z
        float mu   = 10.0f - 9.0f * a;            // MU2 + (MU1-MU2)*a
        float mu_x = -9.0f * a_x, mu_y = -9.0f * a_y;
        float rho  = 1000.0f - 900.0f * a;        // RHO2 + (RHO1-RHO2)*a
        float g2   = a_x * a_x + a_y * a_y + 2.220446049250313e-16f;
        float gm   = sqrtf(g2);
        float curv = -((a_xx + a_yy) / gm -
                       (a_x * a_x * a_xx + a_y * a_y * a_yy + 2.0f * a_x * a_y * a_xy) / (gm * g2));
        const float ONE_WE = 0.049f;   // SIGMA/(RHO_REF*U_REF^2*L_REF)
        const float ONE_FR = 0.49f;    // G*L_REF/U_REF^2
        const float INV_RE_DEN = 1.0f / 500.0f;
        float one_Re   = mu * INV_RE_DEN;
        float one_Re_x = mu_x * INV_RE_DEN;
        float one_Re_y = mu_y * INV_RE_DEN;
        float rr = rho * 1e-3f;        // rho / RHO_REF
        float PDE_m = u_x + v_y;
        float PDE_a = a_t + u * a_x + v * a_y;
        float PDE_u = (u_t + u * u_x + v * u_y) * rr + p_x - ONE_WE * curv * a_x
                      - one_Re * (u_xx + u_yy) - 2.0f * one_Re_x * u_x - one_Re_y * (u_y + v_x);
        float PDE_v = (v_t + u * v_x + v * v_y) * rr + p_y - ONE_WE * curv * a_y
                      - one_Re * (v_xx + v_yy) - rr * ONE_FR
                      - 2.0f * one_Re_y * v_y - one_Re_x * (u_y + v_x);
        int gp = p0 + tid;
        out[gp]            = PDE_m;
        out[NPTS + gp]     = PDE_u;
        out[2 * NPTS + gp] = PDE_v;
        out[3 * NPTS + gp] = PDE_a;
    }
}

extern "C" void kernel_launch(void* const* d_in, const int* in_sizes, int n_in,
                              void* d_out, int out_size, void* d_ws, size_t ws_size,
                              hipStream_t stream) {
    const float* x     = (const float*)d_in[0];
    const float* y     = (const float*)d_in[1];
    const float* t     = (const float*)d_in[2];
    const float* W_in  = (const float*)d_in[3];
    const float* b_in  = (const float*)d_in[4];
    const float* W_h   = (const float*)d_in[5];
    const float* b_h   = (const float*)d_in[6];
    const float* W_out = (const float*)d_in[7];
    const float* b_out = (const float*)d_in[8];
    const float* act   = (const float*)d_in[9];
    half_t* Wp = (half_t*)d_ws;        // 6 * 256 * 256 * 2 = 768 KB
    float* out = (float*)d_out;

    prep_weights<<<(NLAYERS * HP * HP + 255) / 256, 256, 0, stream>>>(W_h, Wp);
    pinn_fused<<<NPTS / TP, 256, SMEM_BYTES, stream>>>(
        x, y, t, W_in, b_in, b_h, W_out, b_out, act, Wp, out);
}
